// FlashAttentionWithRoPE_15839839388339
// MI455X (gfx1250) — matrix operand order repaired and checked by reading, not compiled, not run on hardware
//
#include <hip/hip_runtime.h>

typedef _Float16 h16;
typedef __attribute__((ext_vector_type(16))) _Float16 v16h;
typedef __attribute__((ext_vector_type(8)))  float    v8f;

#define C_DIM  768
#define N_TOK  2304
#define B_SZ   2
#define NH     12
#define HD     64
#define BN_TOK (B_SZ * N_TOK)   // 4608
#define W_IMG  48
#define KT     64               // key tile in flash attention

#define USE_ASYNC_LDS 1

// ---- helpers ---------------------------------------------------------------
__device__ __forceinline__ v16h make_frag(uint4 a, uint4 b) {
    union { v16h v; uint4 q[2]; } u;
    u.q[0] = a; u.q[1] = b;
    return u.v;
}

// 16B global -> LDS copy. CDNA5 async path (ASYNCcnt) with VGPR-roundtrip fallback.
__device__ __forceinline__ void ld16_to_lds(void* lds, const void* g) {
#if USE_ASYNC_LDS
    unsigned int       l = (unsigned int)(size_t)lds;        // low 32 bits = LDS offset
    unsigned long long a = (unsigned long long)(size_t)g;
    asm volatile("global_load_async_to_lds_b128 %0, %1, off"
                 :: "v"(l), "v"(a) : "memory");
#else
    *(uint4*)lds = *(const uint4*)g;
#endif
}
__device__ __forceinline__ void lds_copy_wait() {
#if USE_ASYNC_LDS
    asm volatile("s_wait_asynccnt 0x0" ::: "memory");
#endif
}

// ---- Kernel 1: LayerNorm x3 + [B,C,N] -> [B*N,C] transpose + f16 convert ---
__global__ void ln3_kernel(const float* __restrict__ x,
                           const float* __restrict__ gq, const float* __restrict__ bq,
                           const float* __restrict__ gk, const float* __restrict__ bk,
                           const float* __restrict__ gv, const float* __restrict__ bv,
                           h16* __restrict__ xnq, h16* __restrict__ xnk, h16* __restrict__ xnv) {
    __shared__ float red[256];
    const int row = blockIdx.x;            // b*N + n
    const int b = row / N_TOK, n = row % N_TOK;
    const int tid = threadIdx.x;

    float v[3];
    float s = 0.f, s2 = 0.f;
    for (int i = 0; i < 3; ++i) {
        int c = tid + i * 256;
        float t = x[((size_t)(b * C_DIM + c)) * N_TOK + n];
        v[i] = t; s += t; s2 += t * t;
    }
    red[tid] = s; __syncthreads();
    for (int off = 128; off > 0; off >>= 1) { if (tid < off) red[tid] += red[tid + off]; __syncthreads(); }
    float mu = red[0] * (1.0f / C_DIM);
    __syncthreads();
    red[tid] = s2; __syncthreads();
    for (int off = 128; off > 0; off >>= 1) { if (tid < off) red[tid] += red[tid + off]; __syncthreads(); }
    float var  = red[0] * (1.0f / C_DIM) - mu * mu;
    float rstd = rsqrtf(var + 1e-5f);

    for (int i = 0; i < 3; ++i) {
        int c = tid + i * 256;
        float xn = (v[i] - mu) * rstd;
        size_t o = (size_t)row * C_DIM + c;
        xnq[o] = (h16)(xn * gq[c] + bq[c]);
        xnk[o] = (h16)(xn * gk[c] + bk[c]);
        xnv[o] = (h16)(xn * gv[c] + bv[c]);
    }
}

// ---- Kernel 2: weights fp32 -> f16 ----------------------------------------
__global__ void cvt_w_kernel(const float* __restrict__ wq, const float* __restrict__ wk,
                             const float* __restrict__ wv, const float* __restrict__ wp,
                             h16* q, h16* k, h16* v, h16* p) {
    int i = blockIdx.x * 256 + threadIdx.x;
    if (i < C_DIM * C_DIM) {
        q[i] = (h16)wq[i]; k[i] = (h16)wk[i];
        v[i] = (h16)wv[i]; p[i] = (h16)wp[i];
    }
}

// ---- Kernel 3: WMMA GEMM  D = A[M,K] * W[N,K]^T, 128x128 tile, double-buffered
// mode 0: RoPE -> q_rope [b,h,n,hd] f16
// mode 1: RoPE -> k_rope [b,h,n,hd] f16
// mode 2: ->   vT        [b,h,hd,n] f16
// mode 3: +bias, fp32, transpose -> out [B,C,N]
__global__ void gemm_ep_kernel(const h16* __restrict__ A, const h16* __restrict__ Wt,
                               h16* __restrict__ outh, float* __restrict__ outf,
                               const float* __restrict__ bias, int mode) {
    __shared__ __align__(16) h16 As[2][128][40];   // 32 K + 8 pad
    __shared__ __align__(16) h16 Bs[2][128][40];

    const int tid  = threadIdx.x;
    const int wave = tid >> 5, lane = tid & 31;
    const int lan  = lane & 15, lh = lane >> 4;
    const int m0 = blockIdx.x * 128;
    const int n0 = blockIdx.y * 128;
    const int wm = wave >> 1, wn = wave & 1;       // wave tile: 32 rows x 64 cols

    v8f acc[2][4] = {};

    auto stage = [&](int buf, int k0) {
        for (int i = 0; i < 2; ++i) {
            int idx = tid + i * 256;               // 512 chunks of 8 h16
            int r = idx >> 2, cc = (idx & 3) * 8;
            ld16_to_lds(&As[buf][r][cc], &A [(size_t)(m0 + r) * C_DIM + k0 + cc]);
            ld16_to_lds(&Bs[buf][r][cc], &Wt[(size_t)(n0 + r) * C_DIM + k0 + cc]);
        }
    };

    const int nsteps = C_DIM / 32;
    stage(0, 0);
    for (int s = 0; s < nsteps; ++s) {
        const int buf = s & 1;
        lds_copy_wait();                 // own async copies for buf complete
        __syncthreads();                 // everyone's staging of buf complete;
                                         // also proves compute(s-1) done -> buf^1 free
        if (s + 1 < nsteps) stage(buf ^ 1, (s + 1) * 32);   // overlaps with compute below

        v16h bf[4];
        for (int ni = 0; ni < 4; ++ni) {
            int n = wn * 64 + ni * 16 + lan;
            // B frag: halves [0..7] = K(lh*8..+8), [8..15] = K(16+lh*8..+8)
            bf[ni] = make_frag(*(const uint4*)&Bs[buf][n][lh * 8],
                               *(const uint4*)&Bs[buf][n][16 + lh * 8]);
        }
        for (int mi = 0; mi < 2; ++mi) {
            int m = wm * 32 + mi * 16 + lan;
            // A frag: halves [0..7] = K(lh*8..+8), [8..15] = K(16+lh*8..+8)
            v16h af = make_frag(*(const uint4*)&As[buf][m][lh * 8],
                                *(const uint4*)&As[buf][m][16 + lh * 8]);
            for (int ni = 0; ni < 4; ++ni)
                acc[mi][ni] = __builtin_amdgcn_wmma_f32_16x16x32_f16(
                    false, af, false, bf[ni], (short)0, acc[mi][ni], false, false);
        }
    }

    // epilogue: C/D layout -> element (M = r + 8*lh, N = lan) in VGPR r
    for (int mi = 0; mi < 2; ++mi)
    for (int ni = 0; ni < 4; ++ni)
    for (int r = 0; r < 8; ++r) {
        int m = m0 + wm * 32 + mi * 16 + r + 8 * lh;   // token row
        int c = n0 + wn * 64 + ni * 16 + lan;          // output channel
        float val = acc[mi][ni][r];
        int b = m / N_TOK, nimg = m % N_TOK;
        if (mode <= 1) {
            int h = c >> 6, d = c & 63;
            float pos = (float)((d < 32) ? (nimg / W_IMG) : (nimg % W_IMG));
            // angle = pos * 10000^(-(d%16)/16) = pos * exp(-(d%16) * ln(1e4)/16)
            float ang = pos * __expf(-(float)(d & 15) * 0.57564627324851149f);
            float cs = __cosf(ang), sn = __sinf(ang);
            float other = __shfl_xor(val, 1, 32);      // partner column d^1, same row
            float res = (d & 1) ? (other * sn + val * cs)
                                : (val * cs - other * sn);
            outh[(size_t)((b * NH + h) * N_TOK + nimg) * HD + d] = (h16)res;
        } else if (mode == 2) {
            int h = c >> 6, d = c & 63;
            outh[(size_t)((b * NH + h) * HD + d) * N_TOK + nimg] = (h16)val;
        } else {
            outf[(size_t)(b * C_DIM + c) * N_TOK + nimg] = val + bias[c];
        }
    }
}

// ---- Kernel 4: flash attention (online softmax, WMMA QK^T and PV) ----------
__global__ void flash_kernel(const h16* __restrict__ qr, const h16* __restrict__ kr,
                             const h16* __restrict__ vt, h16* __restrict__ o16) {
    __shared__ __align__(16) h16 ks[KT][72];       // keys  [key][d]
    __shared__ __align__(16) h16 vs[HD][72];       // V^T   [d][key]
    __shared__ __align__(16) h16 ps[8][16][72];    // per-wave P staging [row][key]

    const int tid = threadIdx.x, wave = tid >> 5, lane = tid & 31;
    const int lan = lane & 15, lh = lane >> 4;
    const int q0 = blockIdx.x * 128;
    const int bh = blockIdx.y;                     // b*NH + h
    const size_t base = (size_t)bh * N_TOK * HD;

    // load Q fragments (A-layout, straight from global: row-major [token, d])
    v16h aq[2];
    {
        int m = q0 + wave * 16 + lan;
        const h16* qrow = qr + base + (size_t)m * HD;
        for (int j = 0; j < 2; ++j)
            aq[j] = make_frag(*(const uint4*)&qrow[j * 32 + lh * 8],
                              *(const uint4*)&qrow[j * 32 + 16 + lh * 8]);
    }

    v8f o[4] = {};
    float mrow[8], lrow[8];
    for (int r = 0; r < 8; ++r) { mrow[r] = -1e30f; lrow[r] = 0.f; }
    const float scale = 0.125f;                    // hd^-0.5

    for (int kt = 0; kt < N_TOK / KT; ++kt) {
        // stage K tile and V^T tile asynchronously (each 64x64 f16, 2 chunks/thread)
        for (int i = 0; i < 2; ++i) {
            int idx = tid + i * 256;
            int r = idx >> 3, cc = (idx & 7) * 8;
            ld16_to_lds(&ks[r][cc], &kr[base + (size_t)(kt * KT + r) * HD + cc]);
            ld16_to_lds(&vs[r][cc], &vt[base + (size_t)r * N_TOK + kt * KT + cc]);
        }
        lds_copy_wait();
        __syncthreads();

        // S = Q * K^T   (16 queries x 64 keys per wave)
        v8f s[4] = {};
        for (int nt = 0; nt < 4; ++nt) {
            int key = nt * 16 + lan;
            for (int j = 0; j < 2; ++j) {
                v16h bf = make_frag(*(const uint4*)&ks[key][j * 32 + lh * 8],
                                    *(const uint4*)&ks[key][j * 32 + 16 + lh * 8]);
                s[nt] = __builtin_amdgcn_wmma_f32_16x16x32_f16(
                    false, aq[j], false, bf, (short)0, s[nt], false, false);
            }
        }

        // online softmax per owned row; row lives across 16 lanes of one half
        for (int r = 0; r < 8; ++r) {
            float vmax = -1e30f;
            for (int nt = 0; nt < 4; ++nt) vmax = fmaxf(vmax, s[nt][r]);
            for (int off = 1; off < 16; off <<= 1)
                vmax = fmaxf(vmax, __shfl_xor(vmax, off, 32));
            vmax *= scale;
            float mnew  = fmaxf(mrow[r], vmax);
            float alpha = __expf(mrow[r] - mnew);
            mrow[r] = mnew;
            float psum = 0.f;
            for (int nt = 0; nt < 4; ++nt) {
                float p = __expf(s[nt][r] * scale - mnew);
                psum += p;
                ps[wave][r + 8 * lh][nt * 16 + lan] = (h16)p;
            }
            for (int off = 1; off < 16; off <<= 1)
                psum += __shfl_xor(psum, off, 32);
            lrow[r] = lrow[r] * alpha + psum;
            for (int dt = 0; dt < 4; ++dt) o[dt][r] *= alpha;
        }

        // O += P * V   (P re-read from LDS in A-fragment layout; same wave -> in-order LDS)
        for (int j = 0; j < 2; ++j) {
            v16h pf = make_frag(*(const uint4*)&ps[wave][lan][j * 32 + lh * 8],
                                *(const uint4*)&ps[wave][lan][j * 32 + 16 + lh * 8]);
            for (int dt = 0; dt < 4; ++dt) {
                int d = dt * 16 + lan;
                v16h bf = make_frag(*(const uint4*)&vs[d][j * 32 + lh * 8],
                                    *(const uint4*)&vs[d][j * 32 + 16 + lh * 8]);
                o[dt] = __builtin_amdgcn_wmma_f32_16x16x32_f16(
                    false, pf, false, bf, (short)0, o[dt], false, false);
            }
        }
        __syncthreads();   // single-buffered ks/vs: all waves done before next stage
    }

    // normalize and write to [B*N, C] f16 for the output projection
    const int b = bh / NH, h = bh % NH;
    for (int dt = 0; dt < 4; ++dt)
    for (int r = 0; r < 8; ++r) {
        int m = q0 + wave * 16 + r + 8 * lh;
        int c = h * HD + dt * 16 + lan;
        float val = o[dt][r] / lrow[r];
        o16[(size_t)(b * N_TOK + m) * C_DIM + c] = (h16)val;
    }
}

// ---- host launch -----------------------------------------------------------
extern "C" void kernel_launch(void* const* d_in, const int* in_sizes, int n_in,
                              void* d_out, int out_size, void* d_ws, size_t ws_size,
                              hipStream_t stream) {
    const float* x  = (const float*)d_in[0];
    const float* gq = (const float*)d_in[1];
    const float* bq = (const float*)d_in[2];
    const float* gk = (const float*)d_in[3];
    const float* bk = (const float*)d_in[4];
    const float* gv = (const float*)d_in[5];
    const float* bv = (const float*)d_in[6];
    const float* Wq = (const float*)d_in[7];
    const float* Wk = (const float*)d_in[8];
    const float* Wv = (const float*)d_in[9];
    const float* Wp = (const float*)d_in[10];
    const float* bp = (const float*)d_in[11];
    float* out = (float*)d_out;

    const size_t BNC = (size_t)BN_TOK * C_DIM;
    const size_t CC  = (size_t)C_DIM * C_DIM;
    h16* ws   = (h16*)d_ws;
    h16* xnq  = ws;
    h16* xnk  = xnq + BNC;
    h16* xnv  = xnk + BNC;
    h16* wq16 = xnv + BNC;
    h16* wk16 = wq16 + CC;
    h16* wv16 = wk16 + CC;
    h16* wp16 = wv16 + CC;
    h16* qr   = wp16 + CC;   // [b,h,n,hd]
    h16* krp  = qr + BNC;    // [b,h,n,hd]
    h16* vtp  = krp + BNC;   // [b,h,hd,n]
    h16* o16  = vtp + BNC;   // [b*n, C]

    ln3_kernel<<<BN_TOK, 256, 0, stream>>>(x, gq, bq, gk, bk, gv, bv, xnq, xnk, xnv);
    cvt_w_kernel<<<(C_DIM * C_DIM + 255) / 256, 256, 0, stream>>>(Wq, Wk, Wv, Wp,
                                                                  wq16, wk16, wv16, wp16);
    dim3 ggrid(BN_TOK / 128, C_DIM / 128);
    gemm_ep_kernel<<<ggrid, 256, 0, stream>>>(xnq, wq16, qr, nullptr, nullptr, 0);
    gemm_ep_kernel<<<ggrid, 256, 0, stream>>>(xnk, wk16, krp, nullptr, nullptr, 1);
    gemm_ep_kernel<<<ggrid, 256, 0, stream>>>(xnv, wv16, vtp, nullptr, nullptr, 2);

    dim3 fgrid(N_TOK / 128, B_SZ * NH);
    flash_kernel<<<fgrid, 256, 0, stream>>>(qr, krp, vtp, o16);

    gemm_ep_kernel<<<ggrid, 256, 0, stream>>>(o16, wp16, nullptr, out, bp, 3);
}
